// MultiScaleRetention_37452114821262
// MI455X (gfx1250) — matrix operand order repaired and checked by reading, not compiled, not run on hardware
//
#include <hip/hip_runtime.h>

// ---------------------------------------------------------------------------
// MultiScaleRetention forward for MI455X (gfx1250), wave32 + WMMA bf16.
// B=2, N=2048, E=512, H=8, D=64.  All matmuls on v_wmma_f32_16x16x32_bf16.
// Tile staging in the retention kernel uses GLOBAL_LOAD_ASYNC_TO_LDS_B128
// (ASYNCcnt) -- the CDNA5 async global->LDS path.
// ---------------------------------------------------------------------------

typedef __bf16  v8bf  __attribute__((ext_vector_type(8)));
typedef __bf16  v16bf __attribute__((ext_vector_type(16)));
typedef float   v8f   __attribute__((ext_vector_type(8)));

__device__ __forceinline__ __bf16 f2bf(float f) {
    unsigned int u = __builtin_bit_cast(unsigned int, f);
    unsigned int r = (u + 0x7FFFu + ((u >> 16) & 1u)) >> 16;   // RNE
    return __builtin_bit_cast(__bf16, (unsigned short)r);
}

// A-fragment (16x32 bf16): lane (hb = lane>>4) holds K = hb*8..+8 and 16+hb*8..+8
__device__ __forceinline__ v16bf ld_fragA(const __bf16* rowk, int hb) {
    const __bf16* p = rowk + hb * 8;
    v8bf lo = *(const v8bf*)(p);        // K = hb*8 .. hb*8+7
    v8bf hi = *(const v8bf*)(p + 16);   // K = 16+hb*8 .. +7
    return __builtin_shufflevector(lo, hi, 0,1,2,3,4,5,6,7,8,9,10,11,12,13,14,15);
}

// B-fragment (32x16 bf16): lane holds K = hb*8..+8 and 16+hb*8..+8 of column (lane&15)
__device__ __forceinline__ v16bf ld_fragB(const __bf16* rowk, int hb) {
    const __bf16* p = rowk + hb * 8;
    v8bf lo = *(const v8bf*)(p);
    v8bf hi = *(const v8bf*)(p + 16);
    return __builtin_shufflevector(lo, hi, 0,1,2,3,4,5,6,7,8,9,10,11,12,13,14,15);
}

__device__ __forceinline__ v8f wmma_bf16(v16bf a, v16bf b, v8f c) {
    return __builtin_amdgcn_wmma_f32_16x16x32_bf16(false, a, false, b,
                                                   (short)0, c, false, false);
}

// async global -> LDS copy, 16B per lane, tracked with ASYNCcnt
__device__ __forceinline__ void async_ld16(const void* g, void* lds) {
    unsigned loff = (unsigned)(uintptr_t)lds;   // low 32 bits = LDS offset
    asm volatile("global_load_async_to_lds_b128 %0, %1, off"
                 :: "v"(loff), "v"(g) : "memory");
}
__device__ __forceinline__ void async_wait0() {
    asm volatile("s_wait_asynccnt 0" ::: "memory");
}

// ---------------------------------------------------------------------------
// C = A[4096x512] @ W^T[512x512] + bias  (optional SiLU), fp32 in/out,
// bf16 WMMA with fp32 accumulation.  Block: 256 thr = 8 waves, tile 128x64.
// ---------------------------------------------------------------------------
#define LDK 72   // 64 K + 8 pad (bf16 elements), keeps 16B alignment (144 B rows)

__global__ __launch_bounds__(256)
void msr_gemm512(const float* __restrict__ A, const float* __restrict__ W,
                 const float* __restrict__ bias, float* __restrict__ C,
                 int act_silu) {
    const int t    = threadIdx.x;
    const int wave = t >> 5;
    const int lane = t & 31;
    const int hb   = lane >> 4;
    const int lr   = lane & 15;
    const int m0   = blockIdx.y * 128;
    const int n0   = blockIdx.x * 64;

    __shared__ __align__(16) __bf16 As[128 * LDK];
    __shared__ __align__(16) __bf16 Ws[64 * LDK];

    v8f acc[4] = {{}, {}, {}, {}};

    for (int kb = 0; kb < 8; ++kb) {
        const int k0 = kb * 64;
        // prefetch next k-tile into cache while this one is staged/consumed
        if (kb + 1 < 8) {
            int nk = k0 + 64;
            __builtin_prefetch(A + (size_t)(m0 + (t >> 1)) * 512 + nk + (t & 1) * 32, 0, 0);
            if (t < 128)
                __builtin_prefetch(W + (size_t)(n0 + (t >> 1)) * 512 + nk + (t & 1) * 32, 0, 0);
        }
        // stage A tile 128x64 (2048 float4 / 256 thr = 8 each), fp32 -> bf16
        #pragma unroll
        for (int i = 0; i < 8; ++i) {
            int f = i * 256 + t;
            int row = f >> 4, c4 = (f & 15) * 4;
            float4 d = *(const float4*)(A + (size_t)(m0 + row) * 512 + k0 + c4);
            __bf16* dst = &As[row * LDK + c4];
            dst[0] = f2bf(d.x); dst[1] = f2bf(d.y); dst[2] = f2bf(d.z); dst[3] = f2bf(d.w);
        }
        // stage W tile 64x64 (1024 float4 / 256 thr = 4 each)
        #pragma unroll
        for (int i = 0; i < 4; ++i) {
            int f = i * 256 + t;
            int row = f >> 4, c4 = (f & 15) * 4;
            float4 d = *(const float4*)(W + (size_t)(n0 + row) * 512 + k0 + c4);
            __bf16* dst = &Ws[row * LDK + c4];
            dst[0] = f2bf(d.x); dst[1] = f2bf(d.y); dst[2] = f2bf(d.z); dst[3] = f2bf(d.w);
        }
        __syncthreads();

        v16bf a0 = ld_fragA(&As[(wave * 16 + lr) * LDK + 0],  hb);
        v16bf a1 = ld_fragA(&As[(wave * 16 + lr) * LDK + 32], hb);
        #pragma unroll
        for (int j = 0; j < 4; ++j) {
            v16bf b0 = ld_fragB(&Ws[(j * 16 + lr) * LDK + 0],  hb);
            v16bf b1 = ld_fragB(&Ws[(j * 16 + lr) * LDK + 32], hb);
            acc[j] = wmma_bf16(a0, b0, acc[j]);
            acc[j] = wmma_bf16(a1, b1, acc[j]);
        }
        __syncthreads();
    }

    // epilogue: 16x16 f32 D tile layout: row = hb*8+r, col = lane&15
    #pragma unroll
    for (int j = 0; j < 4; ++j) {
        int col = n0 + j * 16 + lr;
        float bv = bias[col];
        #pragma unroll
        for (int r = 0; r < 8; ++r) {
            int row = m0 + wave * 16 + hb * 8 + r;
            float v = acc[j][r] + bv;
            if (act_silu) v = v * __builtin_amdgcn_rcpf(1.0f + __expf(-v));  // SiLU
            C[(size_t)row * 512 + col] = v;
        }
    }
}

// ---------------------------------------------------------------------------
// Rotary theta-shift + bf16 pack.  q,k -> [bh][n][d] (k scaled 1/sqrt(D));
// v -> transposed [bh][d][n] so retention's B-operand rows are s-contiguous.
// ---------------------------------------------------------------------------
__global__ __launch_bounds__(256)
void msr_pack_rotary(const float* __restrict__ qf, const float* __restrict__ kf,
                     const float* __restrict__ vf,
                     __bf16* __restrict__ qb, __bf16* __restrict__ kb,
                     __bf16* __restrict__ vt) {
    int p     = blockIdx.x * 256 + threadIdx.x;   // 0 .. 4096*256-1
    int ep    = p & 255;                          // pair index within E
    int token = p >> 8;                           // 0 .. 4095
    int h     = ep >> 5;                          // head
    int j     = ep & 31;                          // pair within head (D/2 = 32)
    int n     = token & 2047;
    int b     = token >> 11;

    // thetas = repeat(10000^-linspace(0,1,32), 2); angle = n * theta_j
    float texp  = (float)j * (1.0f / 31.0f);
    float theta = __expf(-texp * 9.210340371976184f);   // ln(10000)
    float s, c;
    __sincosf((float)n * theta, &s, &c);

    size_t base = (size_t)token * 512 + h * 64 + 2 * j;
    float q0 = qf[base], q1 = qf[base + 1];
    float k0 = kf[base], k1 = kf[base + 1];
    float v0 = vf[base], v1 = vf[base + 1];

    float q0r = q0 * c - q1 * s, q1r = q1 * c + q0 * s;
    float k0r = (k0 * c - k1 * s) * 0.125f;             // 1/sqrt(64)
    float k1r = (k1 * c + k0 * s) * 0.125f;

    int bh = b * 8 + h;
    size_t qkb = ((size_t)bh * 2048 + n) * 64 + 2 * j;
    qb[qkb] = f2bf(q0r); qb[qkb + 1] = f2bf(q1r);
    kb[qkb] = f2bf(k0r); kb[qkb + 1] = f2bf(k1r);
    size_t vb = (size_t)bh * 64 * 2048 + (size_t)(2 * j) * 2048 + n;
    vt[vb] = f2bf(v0); vt[vb + 2048] = f2bf(v1);
}

// ---------------------------------------------------------------------------
// Streaming retention: per (bh, 64-query-row block), causal loop over 64-key
// tiles.  Tiles staged with async global->LDS (ASYNCcnt); sim tile never
// leaves LDS.  Epilogue: GroupNorm over d + gate.
// Block: 256 thr = 8 waves; wave w owns msub = w&3, nsubs {2*(w>>2), +1}.
// ---------------------------------------------------------------------------
__global__ __launch_bounds__(256)
void msr_retention(const __bf16* __restrict__ qb, const __bf16* __restrict__ kb,
                   const __bf16* __restrict__ vt, const float* __restrict__ gate,
                   float* __restrict__ gated) {
    const int nb   = blockIdx.x;          // query block (64 rows)
    const int bh   = blockIdx.y;          // 0..15
    const int h    = bh & 7;
    const int b    = bh >> 3;
    const int t    = threadIdx.x;
    const int wave = t >> 5, lane = t & 31;
    const int hb   = lane >> 4, lr = lane & 15;
    const int msub = wave & 3;
    const int ns0  = (wave >> 2) * 2;

    __shared__ __align__(16) __bf16 qs[64 * LDK];
    __shared__ __align__(16) __bf16 ks[64 * LDK];
    __shared__ __align__(16) __bf16 vs[64 * LDK];   // v^T tile: [d][s]
    __shared__ __align__(16) __bf16 sims[64 * LDK]; // decayed sim, bf16 restage
    __shared__ __align__(16) float  retf[64 * 68];

    const int n0 = nb * 64;
    const __bf16* qg = qb + ((size_t)bh * 2048 + n0) * 64;

    // stage q tile 64x64 bf16 via async copy (512 x 16B / 256 thr = 2 each)
    #pragma unroll
    for (int i = 0; i < 2; ++i) {
        int f = i * 256 + t, row = f >> 3, c8 = (f & 7) * 8;
        async_ld16(qg + (size_t)row * 64 + c8, &qs[row * LDK + c8]);
    }

    // gammas = 1 - exp(linspace(ln(1/32), ln(1/512), 8))
    const float glo = -3.4657359027997265f, ghi = -6.238324625039508f;
    float gamma = 1.0f - __expf(glo + (ghi - glo) * ((float)h * (1.0f / 7.0f)));
    float l2g   = __log2f(gamma);

    v8f r0 = {}, r1 = {};   // persistent ret accumulators (two 16x16 d-tiles)

    for (int sb = 0; sb <= nb; ++sb) {
        const int s0 = sb * 64;
        const __bf16* kg = kb + ((size_t)bh * 2048 + s0) * 64;
        const __bf16* vg = vt + (size_t)bh * 64 * 2048 + s0;
        #pragma unroll
        for (int i = 0; i < 2; ++i) {
            int f = i * 256 + t, row = f >> 3, c8 = (f & 7) * 8;
            async_ld16(kg + (size_t)row * 64 + c8,   &ks[row * LDK + c8]);
            async_ld16(vg + (size_t)row * 2048 + c8, &vs[row * LDK + c8]);
        }
        async_wait0();         // this wave's async copies landed in LDS
        __syncthreads();       // everyone's copies visible

        // ---- sim = q @ k^T (K = 64 -> two WMMAs), decay-masked, -> LDS bf16
        v16bf a0 = ld_fragA(&qs[(msub * 16 + lr) * LDK + 0],  hb);
        v16bf a1 = ld_fragA(&qs[(msub * 16 + lr) * LDK + 32], hb);
        #pragma unroll
        for (int jj = 0; jj < 2; ++jj) {
            int nsub = ns0 + jj;
            v16bf b0 = ld_fragB(&ks[(nsub * 16 + lr) * LDK + 0],  hb);
            v16bf b1 = ld_fragB(&ks[(nsub * 16 + lr) * LDK + 32], hb);
            v8f sacc = {};
            sacc = wmma_bf16(a0, b0, sacc);
            sacc = wmma_bf16(a1, b1, sacc);
            int s_g = s0 + nsub * 16 + lr;
            #pragma unroll
            for (int r = 0; r < 8; ++r) {
                int m_g = n0 + msub * 16 + hb * 8 + r;
                float val = (s_g <= m_g)
                          ? sacc[r] * __builtin_amdgcn_exp2f(l2g * (float)(m_g - s_g))
                          : 0.0f;
                sims[(msub * 16 + hb * 8 + r) * LDK + nsub * 16 + lr] = f2bf(val);
            }
        }
        __syncthreads();

        // ---- ret += sim @ v   (A = sim from LDS, B = v^T rows, K = 64)
        v16bf sa0 = ld_fragA(&sims[(msub * 16 + lr) * LDK + 0],  hb);
        v16bf sa1 = ld_fragA(&sims[(msub * 16 + lr) * LDK + 32], hb);
        {
            v16bf b0 = ld_fragB(&vs[((ns0) * 16 + lr) * LDK + 0],  hb);
            v16bf b1 = ld_fragB(&vs[((ns0) * 16 + lr) * LDK + 32], hb);
            r0 = wmma_bf16(sa0, b0, r0);
            r0 = wmma_bf16(sa1, b1, r0);
        }
        {
            v16bf b0 = ld_fragB(&vs[((ns0 + 1) * 16 + lr) * LDK + 0],  hb);
            v16bf b1 = ld_fragB(&vs[((ns0 + 1) * 16 + lr) * LDK + 32], hb);
            r1 = wmma_bf16(sa0, b0, r1);
            r1 = wmma_bf16(sa1, b1, r1);
        }
        __syncthreads();   // before next iteration overwrites ks/vs/sims
    }

    // ---- spill ret tiles to LDS fp32 for row-wise GroupNorm
    #pragma unroll
    for (int r = 0; r < 8; ++r) {
        retf[(msub * 16 + hb * 8 + r) * 68 + (ns0)     * 16 + lr] = r0[r];
        retf[(msub * 16 + hb * 8 + r) * 68 + (ns0 + 1) * 16 + lr] = r1[r];
    }
    __syncthreads();

    // GroupNorm(affine=False) over d (64) per row, then * silu-gate
    if (t < 64) {
        const float* rp = &retf[t * 68];
        float mean = 0.0f;
        #pragma unroll 8
        for (int d = 0; d < 64; ++d) mean += rp[d];
        mean *= (1.0f / 64.0f);
        float var = 0.0f;
        #pragma unroll 8
        for (int d = 0; d < 64; ++d) { float df = rp[d] - mean; var += df * df; }
        var *= (1.0f / 64.0f);
        float inv = rsqrtf(var + 1e-6f);
        int token = b * 2048 + n0 + t;
        const float* gp = gate  + (size_t)token * 512 + h * 64;
        float*       op = gated + (size_t)token * 512 + h * 64;
        #pragma unroll 8
        for (int d = 0; d < 64; ++d)
            op[d] = (rp[d] - mean) * inv * gp[d];
    }
}

// ---------------------------------------------------------------------------
// launch
// ---------------------------------------------------------------------------
extern "C" void kernel_launch(void* const* d_in, const int* in_sizes, int n_in,
                              void* d_out, int out_size, void* d_ws, size_t ws_size,
                              hipStream_t stream) {
    const float* query = (const float*)d_in[0];
    const float* k_in  = (const float*)d_in[1];
    const float* v_in  = (const float*)d_in[2];
    const float* Wq = (const float*)d_in[3];  const float* bq = (const float*)d_in[4];
    const float* Wk = (const float*)d_in[5];  const float* bk = (const float*)d_in[6];
    const float* Wv = (const float*)d_in[7];  const float* bv = (const float*)d_in[8];
    const float* Wg = (const float*)d_in[9];  const float* bg = (const float*)d_in[10];
    const float* Wo = (const float*)d_in[11]; const float* bo = (const float*)d_in[12];

    const size_t SZF = (size_t)4096 * 512;        // tokens * E
    const size_t SZH = (size_t)16 * 2048 * 64;    // bh * N * D

    float*  qf     = (float*)d_ws;
    float*  kf     = qf + SZF;
    float*  vf     = kf + SZF;
    float*  gatef  = vf + SZF;
    float*  gatedf = gatef + SZF;
    __bf16* qb     = (__bf16*)(gatedf + SZF);
    __bf16* kb     = qb + SZH;
    __bf16* vt     = kb + SZH;

    dim3 gg(8, 32);   // 512/64 n-tiles, 4096/128 m-tiles

    msr_gemm512<<<gg, 256, 0, stream>>>(query, Wq, bq, qf, 0);
    msr_gemm512<<<gg, 256, 0, stream>>>(k_in,  Wk, bk, kf, 0);
    msr_gemm512<<<gg, 256, 0, stream>>>(v_in,  Wv, bv, vf, 0);
    msr_gemm512<<<gg, 256, 0, stream>>>(query, Wg, bg, gatef, 1);

    msr_pack_rotary<<<4096, 256, 0, stream>>>(qf, kf, vf, qb, kb, vt);

    msr_retention<<<dim3(32, 16), 256, 0, stream>>>(qb, kb, vt, gatef, gatedf);

    msr_gemm512<<<gg, 256, 0, stream>>>(gatedf, Wo, bo, (float*)d_out, 0);
}
